// SilkFeatureNetPL_87917980549400
// MI455X (gfx1250) — compile-verified, matrix operand order checked by reading, not run on hardware
//
#include <hip/hip_runtime.h>
#include <hip/hip_bf16.h>

// ---------------------------------------------------------------------------
// SilkFeatureNet pipeline for MI455X (gfx1250, wave32, WMMA bf16, TDM).
//   prep_weights : one-time f32 -> bf16 repack of all GEMM B-matrices into
//                  [N][K] row-major so fragments are contiguous 16B spans.
//   frontend     : conv1 -> conv2(k=2 causal) -> tconv(k=4,s=4) -> gates_x,
//                  fused per 64-frame tile, GEMMs on v_wmma_f32_16x16x32_bf16.
//   gru          : persistent single-workgroup recurrence; W_hh slice lives in
//                  registers as bf16 B-fragments; per-step gates_x tile is
//                  DMA'd by the Tensor Data Mover (tensor_load_to_lds) with
//                  double buffering + s_wait_tensorcnt.
// ---------------------------------------------------------------------------

typedef __bf16 v16bf __attribute__((ext_vector_type(16)));
typedef float  v8f   __attribute__((ext_vector_type(8)));

#define B_   16
#define T_   4096
#define T4_  1024
#define FD_  47
#define HID_ 64
#define CH_  256
#define G3_  768

__device__ __forceinline__ float sigmoidf(float x) {
  return 1.0f / (1.0f + __expf(-x));
}

__device__ __forceinline__ v8f wmma_bf16(v16bf a, v16bf b, v8f c) {
  return __builtin_amdgcn_wmma_f32_16x16x32_bf16(false, a, false, b, (short)0, c,
                                                 false, false);
}

// 16x32 bf16 A/B fragment from a row-major [M|N][K] matrix (global or LDS).
// Lane layout (ISA 7.12.2): lane holds K = {kb..kb+7} and {kb+16..kb+23}
// (+8 for lanes 16..31) of row (lane&15) -> two contiguous 16-byte loads.
__device__ __forceinline__ v16bf frag16(const __bf16* row, int kbase, int hlf) {
  union { uint4 u[2]; v16bf v; } r;
  r.u[0] = *reinterpret_cast<const uint4*>(row + kbase + 8 * hlf);
  r.u[1] = *reinterpret_cast<const uint4*>(row + kbase + 16 + 8 * hlf);
  return r.v;
}

// ---------------------------------------------------------------------------
// Weight repack: all B-matrices to bf16 [N][K] row-major.
//   w2b [256][512] : w2b[o][kk] = W2[o, kk%256, kk/256]      (conv2, K=512)
//   wtb [1024][256]: wtb[kt*256+o][i] = Wt[i, o, kt]         (tconv)
//   wihb[768][256] : W_ih as-is                              (gates)
//   whhb[768][256] : W_hh as-is                              (GRU)
// ---------------------------------------------------------------------------
__global__ __launch_bounds__(256)
void prep_weights(const float* __restrict__ W2, const float* __restrict__ Wt,
                  const float* __restrict__ Wih, const float* __restrict__ Whh,
                  __bf16* __restrict__ w2b, __bf16* __restrict__ wtb,
                  __bf16* __restrict__ wihb, __bf16* __restrict__ whhb) {
  const int i = blockIdx.x * 256 + threadIdx.x;      // 0 .. 262143
  if (i < 256 * 512) {
    const int o = i >> 9, kk = i & 511;
    w2b[i] = (__bf16)W2[o * 512 + (kk & 255) * 2 + (kk >> 8)];
  }
  {
    const int n = i >> 8, k = i & 255;
    const int kt = n >> 8, o = n & 255;
    wtb[i] = (__bf16)Wt[(k * 256 + o) * 4 + kt];
  }
  if (i < 768 * 256) {
    wihb[i] = (__bf16)Wih[i];
    whhb[i] = (__bf16)Whh[i];
  }
}

// ---------------------------------------------------------------------------
// Frontend: one workgroup = 16 frame-groups (64 frames) of one batch element.
// ---------------------------------------------------------------------------
__global__ __launch_bounds__(256)
void frontend_kernel(const float* __restrict__ features,
                     const float* __restrict__ W1, const float* __restrict__ b1,
                     const __bf16* __restrict__ w2b, const float* __restrict__ b2,
                     const __bf16* __restrict__ wtb, const float* __restrict__ bt,
                     const __bf16* __restrict__ wihb, const float* __restrict__ b_ih,
                     float* __restrict__ gx) {
  __shared__ float  feat[68 * 48];        // 68 frames x 47 features (padded)
  __shared__ __bf16 xg[17 * 256];         // x[t4-1 .. t4+15], conv1 output
  __shared__ __bf16 act2[16 * 256];       // conv2 output (y)
  __shared__ __bf16 cact[64 * 256];       // tconv output (c), 64 frames

  const int tid  = threadIdx.x;
  const int lane = tid & 31;
  const int wv   = tid >> 5;              // 8 waves
  const int hlf  = lane >> 4;
  const int nloc = lane & 15;
  const int b    = blockIdx.y;
  const int t4_0 = blockIdx.x * 16;       // first frame-group of tile
  const int fbase = 4 * t4_0;             // first frame of tile

  // ---- stage features (with 1-group left halo) into LDS -------------------
  for (int idx = tid; idx < 68 * FD_; idx += 256) {
    const int fl = idx / FD_, f = idx % FD_;
    const int gf = fbase - 4 + fl;
    feat[fl * 48 + f] =
        (gf >= 0) ? features[((size_t)b * T_ + gf) * FD_ + f] : 0.0f;
  }
  __syncthreads();

  // ---- conv1: pointwise 47->64 + tanh (VALU) ------------------------------
  {
    const int fi = tid >> 6;              // frame within group (0..3)
    const int h  = tid & 63;              // hidden unit
    float w1r[FD_];
#pragma unroll
    for (int f = 0; f < FD_; ++f) w1r[f] = W1[h * FD_ + f];
    const float bias = b1[h];
    for (int g = 0; g < 17; ++g) {
      float acc = bias;
      const float* fr = feat + (g * 4 + fi) * 48;
#pragma unroll
      for (int f = 0; f < FD_; ++f) acc += fr[f] * w1r[f];
      float val = tanhf(acc);
      if (t4_0 == 0 && g == 0) val = 0.0f;   // causal left pad (x[-1] = 0)
      xg[g * 256 + fi * 64 + h] = (__bf16)val;
    }
  }
  __syncthreads();

  // ---- conv2: GEMM M=16, K=512 (concat x[j-1],x[j] == contiguous in xg),
  //             N=256, then tanh -> act2 ------------------------------------
  {
    const __bf16* arow = xg + nloc * 256;            // A row j = lane&15
#pragma unroll
    for (int nn = 0; nn < 2; ++nn) {
      const int o = (2 * wv + nn) * 16 + nloc;
      const __bf16* brow = w2b + o * 512;
      v8f acc = 0.0f;
#pragma unroll
      for (int ks = 0; ks < 16; ++ks)
        acc = wmma_bf16(frag16(arow, 32 * ks, hlf),
                        frag16(brow, 32 * ks, hlf), acc);
      const float bias = b2[o];
#pragma unroll
      for (int v = 0; v < 8; ++v)
        act2[(v + 8 * hlf) * 256 + o] = (__bf16)tanhf(acc[v] + bias);
    }
  }
  __syncthreads();

  // ---- tconv: GEMM M=16, K=256, N=1024 (col c = kt*256+o) -> cact ---------
  {
    const __bf16* arow = act2 + nloc * 256;
#pragma unroll
    for (int nn = 0; nn < 8; ++nn) {
      const int c  = (8 * wv + nn) * 16 + nloc;
      const int kt = c >> 8, o = c & 255;
      const __bf16* brow = wtb + c * 256;
      v8f acc = 0.0f;
#pragma unroll
      for (int ks = 0; ks < 8; ++ks)
        acc = wmma_bf16(frag16(arow, 32 * ks, hlf),
                        frag16(brow, 32 * ks, hlf), acc);
      const float bias = bt[o];
#pragma unroll
      for (int v = 0; v < 8; ++v)
        cact[(4 * (v + 8 * hlf) + kt) * 256 + o] = (__bf16)tanhf(acc[v] + bias);
    }
  }
  __syncthreads();

  // ---- gates_x = c @ W_ih^T + b_ih : M=64 (4 tiles), K=256, N=768 ---------
#pragma unroll 1
  for (int nn = 0; nn < 6; ++nn) {
    const int cg = (6 * wv + nn) * 16 + nloc;
    const __bf16* brow = wihb + cg * 256;
    v8f acc[4];
#pragma unroll
    for (int m = 0; m < 4; ++m) acc[m] = 0.0f;
#pragma unroll
    for (int ks = 0; ks < 8; ++ks) {
      const v16bf bb = frag16(brow, 32 * ks, hlf);
#pragma unroll
      for (int m = 0; m < 4; ++m)
        acc[m] = wmma_bf16(frag16(cact + (m * 16 + nloc) * 256, 32 * ks, hlf),
                           bb, acc[m]);
    }
    const float bias = b_ih[cg];
#pragma unroll
    for (int m = 0; m < 4; ++m)
#pragma unroll
      for (int v = 0; v < 8; ++v) {
        const int fl = m * 16 + v + 8 * hlf;          // frame local (0..63)
        gx[((size_t)b * T_ + (fbase + fl)) * G3_ + cg] = acc[m][v] + bias;
      }
  }
}

// ---------------------------------------------------------------------------
// TDM: DMA the 16(batch) x 768(gates) f32 tile for one timestep into LDS.
// D# per ISA ch.8: 2D tensor, data_size=4B, dim0=768 (tile0=768),
// dim1=16 (tile1=16), dim0_stride = T*768 elements.
// This toolchain exposes the 6-arg builtin:
//   (u32x4 g0, i32x8 g1, i32x4 g2, i32x4 g3, i32x8 pad, i32 cpol)
// ---------------------------------------------------------------------------
__device__ __forceinline__ void tdm_load_gates(const float* src, void* lds_dst) {
  typedef unsigned int u32x4 __attribute__((ext_vector_type(4)));
  typedef int          i32x4 __attribute__((ext_vector_type(4)));
  typedef int          i32x8 __attribute__((ext_vector_type(8)));
  const unsigned long long ga = (unsigned long long)(uintptr_t)src;
  const unsigned int ldsOff   = (unsigned int)(uintptr_t)lds_dst;  // LDS byte offset
  const u32x4 g0 = {
      1u,                                             // count=1, user mode
      ldsOff,                                         // lds_addr
      (unsigned int)ga,                               // global_addr[31:0]
      ((unsigned int)(ga >> 32) & 0x01FFFFFFu) | (2u << 30)  // addr[56:32]|type=2
  };
  const i32x8 g1 = {
      0x00020000,                // data_size=2 (4B), mask=0, no pad/iter/barrier
      (int)(768u << 16),         // tensor_dim0[15:0]=768
      (int)(16u  << 16),         // tensor_dim0[31:16]=0 | tensor_dim1[15:0]=16
      (int)(768u << 16),         // tensor_dim1[31:16]=0 | tile_dim0=768
      16,                        // tile_dim1=16, tile_dim2=0
      (int)(T_ * G3_),           // tensor_dim0_stride[31:0] = 3145728
      0, 0                       // stride hi / dim1_stride (unused, 2D)
  };
  const i32x4 z4 = {0, 0, 0, 0};
  const i32x8 z8 = {0, 0, 0, 0, 0, 0, 0, 0};
  __builtin_amdgcn_tensor_load_to_lds(g0, g1, z4, z4, z8, 0);
}

// ---------------------------------------------------------------------------
// GRU: one persistent workgroup, 16 waves. Wave w owns channels [16w,16w+16).
// ---------------------------------------------------------------------------
__global__ __launch_bounds__(512)
void gru_kernel(const float* __restrict__ gx,
                const __bf16* __restrict__ whhb,
                const float* __restrict__ b_hh,
                float* __restrict__ out) {
  __shared__ __bf16 hb[2][16 * 256];       // h double buffer   (16 KB)
  __shared__ float  gxl[2][16 * G3_];      // gates_x double buffer (96 KB, TDM dst)

  const int tid  = threadIdx.x;
  const int lane = tid & 31;
  const int wv   = tid >> 5;               // 16 waves
  const int hlf  = lane >> 4;
  const int nloc = lane & 15;
  const int ch   = 16 * wv + nloc;         // owned channel (B/D column)

  // W_hh B-fragments for gates r,z,n (K=256 -> 8 k-steps each), in registers.
  v16bf Bw[3][8];
#pragma unroll
  for (int g = 0; g < 3; ++g) {
    const __bf16* wr = whhb + (size_t)(g * 256 + ch) * 256;
#pragma unroll
    for (int ks = 0; ks < 8; ++ks) Bw[g][ks] = frag16(wr, 32 * ks, hlf);
  }
  const float bhr = b_hh[ch], bhz = b_hh[256 + ch], bhn = b_hh[512 + ch];

  // h = 0; DMA gates for t=0.
  v8f hreg = 0.0f;
  for (int i = tid; i < 16 * 256; i += 512) hb[0][i] = (__bf16)0.0f;
  if (wv == 0) {
    tdm_load_gates(gx, &gxl[0][0]);
    __builtin_amdgcn_s_wait_tensorcnt(0);
  }
  __syncthreads();

  for (int t = 0; t < T_; ++t) {
    // gates for this step (from LDS tile placed by the TDM)
    const float* gl = gxl[t & 1];
    float xgt[3][8];
#pragma unroll
    for (int v = 0; v < 8; ++v) {
      const int bM = v + 8 * hlf;
      xgt[0][v] = gl[bM * G3_ + ch];
      xgt[1][v] = gl[bM * G3_ + 256 + ch];
      xgt[2][v] = gl[bM * G3_ + 512 + ch];
    }
    // kick off DMA for step t+1 while we do this step's math
    if (wv == 0 && t + 1 < T_)
      tdm_load_gates(gx + (size_t)(t + 1) * G3_, &gxl[(t + 1) & 1][0]);

    // gh = h @ W_hh^T  (M=16 batches, N=16 owned channels x 3 gates, K=256)
    v8f ar = 0.0f, az = 0.0f, an = 0.0f;
    const __bf16* hcur = hb[t & 1] + nloc * 256;
#pragma unroll
    for (int ks = 0; ks < 8; ++ks) {
      const v16bf a = frag16(hcur, 32 * ks, hlf);
      ar = wmma_bf16(a, Bw[0][ks], ar);
      az = wmma_bf16(a, Bw[1][ks], az);
      an = wmma_bf16(a, Bw[2][ks], an);
    }

    // elementwise GRU update; h: f32 in registers, bf16 copy to LDS for t+1
    __bf16* hnext = hb[(t + 1) & 1];
#pragma unroll
    for (int v = 0; v < 8; ++v) {
      const float r  = sigmoidf(xgt[0][v] + ar[v] + bhr);
      const float z  = sigmoidf(xgt[1][v] + az[v] + bhz);
      const float n  = tanhf(xgt[2][v] + r * (an[v] + bhn));
      const float hv = (1.0f - z) * n + z * hreg[v];
      hreg[v] = hv;
      const int bM = v + 8 * hlf;
      hnext[bM * 256 + ch] = (__bf16)hv;
      out[((size_t)bM * T_ + t) * CH_ + ch] = hv;
    }
    if (wv == 0) __builtin_amdgcn_s_wait_tensorcnt(0);  // t+1 gates landed
    __syncthreads();                                    // hnext + gxl visible
  }
}

// ---------------------------------------------------------------------------
extern "C" void kernel_launch(void* const* d_in, const int* in_sizes, int n_in,
                              void* d_out, int out_size, void* d_ws, size_t ws_size,
                              hipStream_t stream) {
  const float* features = (const float*)d_in[0];
  const float* W1   = (const float*)d_in[1];
  const float* b1   = (const float*)d_in[2];
  const float* W2   = (const float*)d_in[3];
  const float* b2   = (const float*)d_in[4];
  const float* Wt   = (const float*)d_in[5];
  const float* bt   = (const float*)d_in[6];
  const float* W_ih = (const float*)d_in[7];
  const float* W_hh = (const float*)d_in[8];
  const float* b_ih = (const float*)d_in[9];
  const float* b_hh = (const float*)d_in[10];

  // workspace layout
  char* ws = (char*)d_ws;
  float* gx = (float*)ws;                          // B*T*768 f32 = 192 MiB
  size_t off = (size_t)B_ * T_ * G3_ * sizeof(float);
  __bf16* w2b  = (__bf16*)(ws + off); off += (size_t)256 * 512 * 2;
  __bf16* wtb  = (__bf16*)(ws + off); off += (size_t)1024 * 256 * 2;
  __bf16* wihb = (__bf16*)(ws + off); off += (size_t)768 * 256 * 2;
  __bf16* whhb = (__bf16*)(ws + off);

  prep_weights<<<1024, 256, 0, stream>>>(W2, Wt, W_ih, W_hh, w2b, wtb, wihb, whhb);

  dim3 gridF(T4_ / 16, B_);
  frontend_kernel<<<gridF, 256, 0, stream>>>(features, W1, b1, w2b, b2, wtb, bt,
                                             wihb, b_ih, gx);
  gru_kernel<<<1, 512, 0, stream>>>(gx, whhb, b_hh, (float*)d_out);
}
